// SpriteAssembler_79568564126196
// MI455X (gfx1250) — matrix operand order repaired and checked by reading, not compiled, not run on hardware
//
#include <hip/hip_runtime.h>

// ---------------------------------------------------------------------------
// SpriteAssembler: depth-ordered log-space alpha compositing on MI455X.
//   masks[b,p,i] = normalize_i( s_i * 2^( sum_j log2(1-s_j) * w[b,i,j] ) )
// einsum -> V_WMMA_F32_16X16X32_F16 (A = log2(1-s) tile, B = w^T, C = inline 0)
// exp2(log2 s + occ) == s * exp2(occ): 2 transcendentals/element, no guards.
// Per-wave LDS tiles + in-order wave-wide DS => no workgroup barriers in the
// main loop (wave_barrier scheduling fences only); waves drift to hide HBM.
// ---------------------------------------------------------------------------

typedef __attribute__((ext_vector_type(16))) _Float16 v16h;
typedef __attribute__((ext_vector_type(8)))  _Float16 v8h;
typedef __attribute__((ext_vector_type(4)))  _Float16 v4h;
typedef __attribute__((ext_vector_type(8)))  float    v8f;

#define N_SPR       64
#define THREADS     256
#define WAVES       8
#define PX_PER_WAVE 16
#define PX_PER_IT   128          // WAVES * PX_PER_WAVE
#define ITERS       4
#define PX_PER_BLK  512          // PX_PER_IT * ITERS

#define HSTRIDE     72           // f16 row stride (64 + 8 pad), 144B = 16B-multiple
#define FSTRIDE     68           // f32 row stride (64 + 4 pad), 272B = 16B-multiple
#define EPS_C       1e-6f

union AFrag { v16h v; v8h h[2]; };

__global__ __launch_bounds__(THREADS)
void sprite_composite_wmma(const float* __restrict__ shapes,
                           const float* __restrict__ depths,
                           float* __restrict__ out,
                           int hw_per_batch)
{
    __shared__ alignas(16) _Float16 lds_w[N_SPR * HSTRIDE];                 // w[i][j] as f16 (shared)
    __shared__ alignas(16) _Float16 lds_m[WAVES][PX_PER_WAVE * HSTRIDE];    // log2(1-s) f16 (per-wave)
    __shared__ alignas(16) float    lds_s[WAVES][PX_PER_WAVE * FSTRIDE];    // clipped s / unnorm (per-wave)

    const int tid  = threadIdx.x;
    const int lane = tid & 31;
    const int wid  = tid >> 5;

    const int block_px = blockIdx.x * PX_PER_BLK;          // 512 px, single batch
    const int batch    = block_px / hw_per_batch;
    const float* dptr  = depths + batch * N_SPR;

    // ---- occlusion matrix w[i][j] = tanh(relu(d_j - d_i)), f16, row i-major ----
    for (int e = tid; e < N_SPR * N_SPR; e += THREADS) {
        const int i = e >> 6, j = e & 63;
        const float dd = dptr[j] - dptr[i];
        lds_w[i * HSTRIDE + j] = (_Float16)tanhf(dd > 0.0f ? dd : 0.0f);
    }
    __syncthreads();   // only cross-wave dependency in the kernel (lds_w build)

    // ---- loop-invariant B fragments: b[n][k], K = 32k + 16*(lane>>4) + 2v,2v+1,
    //      Ncol = 16n + (lane&15); element = w^T[K][N] = w[N][K] -> contiguous row read.
    v16h bfrag[4][2];
    {
        const int ncol  = lane & 15;
        const int khalf = (lane >> 4) << 4;                // 0 or 16
#pragma unroll
        for (int n = 0; n < 4; ++n)
#pragma unroll
            for (int k = 0; k < 2; ++k) {
                const _Float16* p = &lds_w[(n * 16 + ncol) * HSTRIDE + k * 32 + khalf];
                AFrag u;
                u.h[0] = *(const v8h*)(p);
                u.h[1] = *(const v8h*)(p + 8);
                bfrag[n][k] = u.v;
            }
    }

    for (int it = 0; it < ITERS; ++it) {
        const int wave_px  = block_px + it * PX_PER_IT + wid * PX_PER_WAVE;
        const float* tile  = shapes + (size_t)wave_px * N_SPR;
        float*       otile = out    + (size_t)wave_px * N_SPR;
        _Float16* lm = lds_m[wid];
        float*    ls = lds_s[wid];

        // per-wave LDS reuse fence (compiler scheduling only; DS is in-order
        // wave-wide, so no hardware barrier is required between phases)
        __builtin_amdgcn_wave_barrier();

        // prefetch next iteration's 4KB tile: one 128B line per lane
        if (it + 1 < ITERS)
            __builtin_prefetch((const char*)(tile + PX_PER_IT * N_SPR) + lane * 128, 0, 0);

        // ---- load 16px x 64 f32 (coalesced b128), clip + single log2, stage to LDS ----
        const float4* src = (const float4*)tile;
#pragma unroll
        for (int t = 0; t < 8; ++t) {
            const int f  = t * 32 + lane;                  // float4 index in tile [0,256)
            float4 s4 = src[f];
            const int px = f >> 4, ch = (f & 15) << 2;
            s4.x = fminf(fmaxf(s4.x, EPS_C), 1.0f - EPS_C);
            s4.y = fminf(fmaxf(s4.y, EPS_C), 1.0f - EPS_C);
            s4.z = fminf(fmaxf(s4.z, EPS_C), 1.0f - EPS_C);
            s4.w = fminf(fmaxf(s4.w, EPS_C), 1.0f - EPS_C);
            *(float4*)&ls[px * FSTRIDE + ch] = s4;         // clipped s, f32
            v4h hm;                                        // log2(1-s), f16
            hm[0] = (_Float16)__log2f(1.0f - s4.x);
            hm[1] = (_Float16)__log2f(1.0f - s4.y);
            hm[2] = (_Float16)__log2f(1.0f - s4.z);
            hm[3] = (_Float16)__log2f(1.0f - s4.w);
            *(v4h*)&lm[px * HSTRIDE + ch] = hm;
        }
        __builtin_amdgcn_wave_barrier();

        // ---- A fragments (contiguous K order matching B: lane-half K 0-15 vs 16-31)
        const int arow = lane & 15;
        const int ak0  = (lane >> 4) << 4;                 // 0 or 16
        AFrag ua0, ua1;
        {
            const _Float16* p0 = &lm[arow * HSTRIDE + ak0];
            ua0.h[0] = *(const v8h*)(p0);
            ua0.h[1] = *(const v8h*)(p0 + 8);
            const _Float16* p1 = &lm[arow * HSTRIDE + 32 + ak0];
            ua1.h[0] = *(const v8h*)(p1);
            ua1.h[1] = *(const v8h*)(p1 + 8);
        }

        // ---- D = A*B (C = inline 0); unnorm = s * exp2(D), raw v_exp_f32 ----
        const int crow = (lane >> 4) << 3;                 // M offset 0 or 8
        const int ncol = lane & 15;
#pragma unroll
        for (int n = 0; n < 4; ++n) {
            v8f zero = {};
            v8f acc = __builtin_amdgcn_wmma_f32_16x16x32_f16(
                          false, ua0.v, false, bfrag[n][0], (short)0, zero, false, false);
            acc = __builtin_amdgcn_wmma_f32_16x16x32_f16(
                          false, ua1.v, false, bfrag[n][1], (short)0, acc, false, false);
#pragma unroll
            for (int r = 0; r < 8; ++r) {
                float* cell = &ls[(crow + r) * FSTRIDE + n * 16 + ncol];
                *cell = *cell * __builtin_amdgcn_exp2f(acc[r]);
            }
        }
        __builtin_amdgcn_wave_barrier();

        // ---- normalize: 2 lanes per pixel, shfl_xor partner sum, coalesced b128 stores
        {
            const int P  = lane >> 1;
            const int hh = (lane & 1) << 5;                // column 0 or 32
            float4 vbuf[8];
            float sum = 0.0f;
#pragma unroll
            for (int q = 0; q < 8; ++q) {
                vbuf[q] = *(const float4*)&ls[P * FSTRIDE + hh + q * 4];
                sum += vbuf[q].x + vbuf[q].y + vbuf[q].z + vbuf[q].w;
            }
            sum += __shfl_xor(sum, 1, 32);
            const float inv = __builtin_amdgcn_rcpf(sum + EPS_C);
            float4* dst = (float4*)(otile + (size_t)P * N_SPR + hh);
#pragma unroll
            for (int q = 0; q < 8; ++q) {
                float4 v = vbuf[q];
                v.x *= inv; v.y *= inv; v.z *= inv; v.w *= inv;
                dst[q] = v;
            }
        }
    }
}

extern "C" void kernel_launch(void* const* d_in, const int* in_sizes, int n_in,
                              void* d_out, int out_size, void* d_ws, size_t ws_size,
                              hipStream_t stream) {
    (void)n_in; (void)out_size; (void)d_ws; (void)ws_size;
    const float* shapes = (const float*)d_in[0];   // [B,H,W,64] f32
    const float* depths = (const float*)d_in[1];   // [B,64]     f32
    float* out = (float*)d_out;                    // [B,H,W,64] f32

    const int total_px     = in_sizes[0] / N_SPR;          // B*H*W = 524288
    const int batches      = in_sizes[1] / N_SPR;          // 8
    const int hw_per_batch = total_px / batches;           // 65536
    const int blocks       = total_px / PX_PER_BLK;        // 1024

    sprite_composite_wmma<<<blocks, THREADS, 0, stream>>>(shapes, depths, out, hw_per_batch);
}
